// Transformer_37297495998565
// MI455X (gfx1250) — matrix operand order repaired and checked by reading, not compiled, not run on hardware
//
#include <hip/hip_runtime.h>
#include <hip/hip_bf16.h>

// ---------------------------------------------------------------------------
// Transformer forward (2-layer, pre-LN) for gfx1250 using bf16 WMMA,
// async global->LDS staging, and permlane16 cross-lane reductions.
// ---------------------------------------------------------------------------

typedef __bf16 bf16;
typedef __attribute__((ext_vector_type(16))) __bf16 v16bf;
typedef __attribute__((ext_vector_type(8)))  float  v8f;

#define L_ 2
#define B_ 2
#define S_ 2048
#define D_ 768
#define H_ 12
#define K_ 64
#define F_ 3072
#define M_ (B_ * S_)          // 4096 rows

// ---------------------------------------------------------------------------
// CDNA5 feature probes (compile-safe guards)
// ---------------------------------------------------------------------------
#if __has_builtin(__builtin_amdgcn_global_load_async_to_lds_b128)
#define HAVE_ASYNC_LDS 1
#else
#define HAVE_ASYNC_LDS 0
#endif

#if __has_builtin(__builtin_amdgcn_permlane16)
#define HAVE_PERMLANE16 1
#else
#define HAVE_PERMLANE16 0
#endif

#if HAVE_ASYNC_LDS
typedef int v4i __attribute__((vector_size(16)));
typedef __attribute__((address_space(1))) v4i g_v4i;   // global (AS1)
typedef __attribute__((address_space(3))) v4i l_v4i;   // LDS (AS3)
// 16-byte global -> LDS async copy (per-lane), tracked by ASYNCcnt.
static __device__ __forceinline__ void async_copy_b128(const void* g, void* l) {
  __builtin_amdgcn_global_load_async_to_lds_b128((g_v4i*)g, (l_v4i*)l, 0, 0);
}
static __device__ __forceinline__ void wait_async0() {
#if __has_builtin(__builtin_amdgcn_s_wait_asynccnt)
  __builtin_amdgcn_s_wait_asynccnt(0);
#else
  asm volatile("s_wait_asynccnt 0" ::: "memory");
#endif
}
#endif

// xor-butterfly lane select tables for v_permlane16 (within 16-lane rows)
static constexpr unsigned psel(int xm, int base) {
  unsigned r = 0;
  for (int i = 0; i < 8; ++i)
    r |= (unsigned)(((base + i) ^ xm) & 15) << (4 * i);
  return r;
}

template <int XM>
static __device__ __forceinline__ float lanexor16(float v) {
#if HAVE_PERMLANE16
  union { float f; unsigned u; } c;
  c.f = v;
  constexpr unsigned s0 = psel(XM, 0);
  constexpr unsigned s1 = psel(XM, 8);
  c.u = __builtin_amdgcn_permlane16(c.u, c.u, s0, s1, false, false);
  return c.f;
#else
  return __shfl_xor(v, XM, 32);
#endif
}

static __device__ __forceinline__ float redmax16(float t) {
  t = fmaxf(t, lanexor16<1>(t));
  t = fmaxf(t, lanexor16<2>(t));
  t = fmaxf(t, lanexor16<4>(t));
  t = fmaxf(t, lanexor16<8>(t));
  return t;
}
static __device__ __forceinline__ float redsum16(float t) {
  t += lanexor16<1>(t);
  t += lanexor16<2>(t);
  t += lanexor16<4>(t);
  t += lanexor16<8>(t);
  return t;
}

// ---------------------------------------------------------------------------
// WMMA helpers (wave32 layouts per CDNA5 ISA 7.12.2)
// ---------------------------------------------------------------------------
static __device__ __forceinline__ v8f wmma_bf16(v16bf a, v16bf b, v8f c) {
  return __builtin_amdgcn_wmma_f32_16x16x32_bf16(
      /*neg_a=*/false, a, /*neg_b=*/false, b,
      /*c_mod=*/(short)0, c, /*reuse_a=*/false, /*reuse_b=*/false);
}

static __device__ __forceinline__ void put2(v16bf& f, int p, unsigned u) {
  union { unsigned u; bf16 h[2]; } cv; cv.u = u;
  f[2 * p]     = cv.h[0];
  f[2 * p + 1] = cv.h[1];
}

// A-fragment 16x32 bf16 from LDS, row-major [m][k].
static __device__ __forceinline__ v16bf frag_a_lds(const bf16* sm, int row0,
                                                   int stride, int lane) {
  v16bf a;
  int m  = row0 + (lane & 15);
  int kb = (lane >> 4) << 3;
  const bf16* rp = sm + m * stride;
#pragma unroll
  for (int p = 0; p < 8; ++p) {
    int k = (p < 4) ? (kb + 2 * p) : (16 + kb + 2 * (p - 4));
    put2(a, p, *(const unsigned*)(rp + k));
  }
  return a;
}

// A-fragment 16x32 bf16 directly from global (rowStride in elements).
static __device__ __forceinline__ v16bf frag_a_global(const bf16* base,
                                                      int rowStride, int koff,
                                                      int lane) {
  v16bf a;
  int m  = lane & 15;
  int kb = (lane >> 4) << 3;
  const bf16* rp = base + (size_t)m * rowStride + koff;
#pragma unroll
  for (int p = 0; p < 8; ++p) {
    int k = (p < 4) ? (kb + 2 * p) : (16 + kb + 2 * (p - 4));
    put2(a, p, *(const unsigned*)(rp + k));
  }
  return a;
}

// A-fragment built from a per-wave f32 LDS scratch (softmax P tile).
static __device__ __forceinline__ v16bf frag_a_f32lds(const float* sm,
                                                      int stride, int lane) {
  v16bf a;
  int m  = lane & 15;
  int kb = (lane >> 4) << 3;
  const float* rp = sm + m * stride;
#pragma unroll
  for (int p = 0; p < 8; ++p) {
    int k = (p < 4) ? (kb + 2 * p) : (16 + kb + 2 * (p - 4));
    a[2 * p]     = (bf16)rp[k];
    a[2 * p + 1] = (bf16)rp[k + 1];
  }
  return a;
}

// B-fragment 32x16 bf16 from LDS stored as [n][k] (k contiguous).
static __device__ __forceinline__ v16bf frag_b_lds(const bf16* sm, int col0,
                                                   int stride, int koff,
                                                   int lane) {
  v16bf b;
  int n  = col0 + (lane & 15);
  int kb = (lane >> 4) << 3;
  const bf16* rp = sm + n * stride + koff;
#pragma unroll
  for (int p = 0; p < 8; ++p) {
    int k = (p < 4) ? (kb + 2 * p) : (16 + kb + 2 * (p - 4));
    put2(b, p, *(const unsigned*)(rp + k));
  }
  return b;
}

static __device__ __forceinline__ float gelu_tanh(float v) {
  float c = 0.7978845608028654f * (v + 0.044715f * v * v * v);
  return 0.5f * v * (1.0f + tanhf(c));
}

// ---------------------------------------------------------------------------
// Elementwise utility kernels
// ---------------------------------------------------------------------------
__global__ void cvt_f32_bf16(const float* __restrict__ in,
                             bf16* __restrict__ out, int n) {
  for (int i = blockIdx.x * blockDim.x + threadIdx.x; i < n;
       i += gridDim.x * blockDim.x)
    out[i] = (bf16)in[i];
}

__global__ void copy_f32(const float* __restrict__ in, float* __restrict__ out,
                         int n) {
  for (int i = blockIdx.x * blockDim.x + threadIdx.x; i < n;
       i += gridDim.x * blockDim.x)
    out[i] = in[i];
}

// ---------------------------------------------------------------------------
// LayerNorm: one block (256 thr) per row of D_=768. OUTF=1 -> f32 out.
// ---------------------------------------------------------------------------
template <int OUTF>
__global__ __launch_bounds__(256) void layernorm_k(
    const float* __restrict__ x, const float* __restrict__ sc,
    const float* __restrict__ bi, bf16* __restrict__ ob,
    float* __restrict__ of) {
  __shared__ float red[2][8];
  const int row = blockIdx.x;
  const int t   = threadIdx.x;
  const float* xr = x + (size_t)row * D_;
  float v0 = xr[t], v1 = xr[t + 256], v2 = xr[t + 512];
  float sum = v0 + v1 + v2;
  float sq  = v0 * v0 + v1 * v1 + v2 * v2;
#pragma unroll
  for (int off = 1; off < 32; off <<= 1) {
    sum += __shfl_xor(sum, off, 32);
    sq  += __shfl_xor(sq, off, 32);
  }
  int lane = t & 31, w = t >> 5;
  if (lane == 0) { red[0][w] = sum; red[1][w] = sq; }
  __syncthreads();
  float ts = 0.f, tq = 0.f;
#pragma unroll
  for (int i = 0; i < 8; ++i) { ts += red[0][i]; tq += red[1][i]; }
  const float mean = ts * (1.0f / D_);
  const float var  = tq * (1.0f / D_) - mean * mean;
  const float rstd = rsqrtf(var + 1e-5f);
#pragma unroll
  for (int j = 0; j < 3; ++j) {
    int c   = t + j * 256;
    float v = (j == 0) ? v0 : (j == 1) ? v1 : v2;
    float y = (v - mean) * rstd * sc[c] + bi[c];
    if (OUTF) of[(size_t)row * D_ + c] = y;
    else      ob[(size_t)row * D_ + c] = (bf16)y;
  }
}

// ---------------------------------------------------------------------------
// Tiled WMMA GEMM: C[M,N] = A[M,K] * W[K,N] (+ bias, + residual / GELU)
// Block: 256 thr = 8 waves (4 along M x 2 along N); tile 128x64, K-step 32.
// A tile staged via async global->LDS b128 copies when available.
// ---------------------------------------------------------------------------
#define TM 128
#define TN 64
#define TKK 32
#define LDA_S 40   // 80-byte rows: 16B-aligned chunks, conflict-free frag reads
#define LDB_S 34

enum { EP_BF16_BIAS = 0, EP_F32_BIAS_RES = 1, EP_BF16_BIAS_GELU = 2 };

template <int MODE>
__global__ __launch_bounds__(256, 2) void gemm_bf16_wmma(
    const bf16* __restrict__ A, const bf16* __restrict__ W,
    const float* __restrict__ bias, const float* __restrict__ res,
    bf16* __restrict__ outb, float* __restrict__ outf, int Ndim, int Kdim) {
  __shared__ alignas(16) bf16 As[TM * LDA_S];
  __shared__ alignas(16) bf16 Bs[TN * LDB_S];
  const int lane = threadIdx.x & 31, wave = threadIdx.x >> 5;
  const int wm = wave & 3, wn = wave >> 2;
  const int m0 = blockIdx.y * TM, n0 = blockIdx.x * TN;

  v8f acc[2][2] = {};

  for (int k0 = 0; k0 < Kdim; k0 += TKK) {
    __syncthreads();
    {
      // ---- stage A tile 128x32 (row-major, 16B-aligned padded rows) ----
      int idx = threadIdx.x * 8;
#pragma unroll
      for (int rep = 0; rep < 2; ++rep, idx += 2048) {
        int r = idx >> 5, c = idx & 31;
        const bf16* gp = A + (size_t)(m0 + r) * Kdim + k0 + c;
        bf16* sp = &As[r * LDA_S + c];
#if HAVE_ASYNC_LDS
        async_copy_b128(gp, sp);       // global_load_async_to_lds_b128
#else
        union { float4 v; unsigned u[4]; } t;
        t.v = *(const float4*)gp;
        unsigned* su = (unsigned*)sp;
        su[0] = t.u[0]; su[1] = t.u[1]; su[2] = t.u[2]; su[3] = t.u[3];
#endif
      }
      // ---- stage B tile 32x64, transposed into Bs[n][k] ----
      int idy = threadIdx.x * 8;
      int kk = idy >> 6, nn = idy & 63;
      union { float4 v; bf16 h[8]; } tb;
      tb.v = *(const float4*)(W + (size_t)(k0 + kk) * Ndim + n0 + nn);
#pragma unroll
      for (int j = 0; j < 8; ++j) Bs[(nn + j) * LDB_S + kk] = tb.h[j];
      // prefetch next K tile (global_prefetch_b8)
      if (k0 + TKK < Kdim) {
        int r0 = (threadIdx.x * 8) >> 5, c0 = (threadIdx.x * 8) & 31;
        __builtin_prefetch(A + (size_t)(m0 + r0) * Kdim + k0 + TKK + c0, 0, 1);
        __builtin_prefetch(W + (size_t)(k0 + TKK + kk) * Ndim + n0 + nn, 0, 1);
      }
    }
#if HAVE_ASYNC_LDS
    wait_async0();                     // s_wait_asynccnt 0
#endif
    __syncthreads();
    // ---- 4 WMMAs per wave per K-step ----
    v16bf a0 = frag_a_lds(As, wm * 32 + 0,  LDA_S, lane);
    v16bf a1 = frag_a_lds(As, wm * 32 + 16, LDA_S, lane);
    v16bf b0 = frag_b_lds(Bs, wn * 32 + 0,  LDB_S, 0, lane);
    v16bf b1 = frag_b_lds(Bs, wn * 32 + 16, LDB_S, 0, lane);
    acc[0][0] = wmma_bf16(a0, b0, acc[0][0]);
    acc[0][1] = wmma_bf16(a0, b1, acc[0][1]);
    acc[1][0] = wmma_bf16(a1, b0, acc[1][0]);
    acc[1][1] = wmma_bf16(a1, b1, acc[1][1]);
  }

  // ---- epilogue ----
  const int nl = lane & 15, hs = lane >> 4;
#pragma unroll
  for (int i = 0; i < 2; ++i)
#pragma unroll
    for (int j = 0; j < 2; ++j)
#pragma unroll
      for (int r = 0; r < 8; ++r) {
        int m = m0 + wm * 32 + i * 16 + r + (hs << 3);
        int n = n0 + wn * 32 + j * 16 + nl;
        float v = acc[i][j][r] + bias[n];
        size_t o = (size_t)m * Ndim + n;
        if (MODE == EP_F32_BIAS_RES)        outf[o] = res[o] + v;
        else if (MODE == EP_BF16_BIAS_GELU) outb[o] = (bf16)gelu_tanh(v);
        else                                outb[o] = (bf16)v;
      }
}

// ---------------------------------------------------------------------------
// Flash attention: block = 4 waves (128 thr) = 64 query rows of one (b,h).
// K tile staged with async b128 copies; V staged transposed; online softmax
// with permlane16 (VALU) row reductions.
// ---------------------------------------------------------------------------
#define ATT_LDK 72   // Ks row stride ([key][d], 144B rows: 16B-aligned)
#define ATT_LDV 34   // Vs row stride ([d][key], key-contiguous, pad)
#define ATT_LDP 33   // per-wave P scratch stride (f32)

__global__ __launch_bounds__(128, 2) void attn_flash(
    const bf16* __restrict__ Q, const bf16* __restrict__ Kt,
    const bf16* __restrict__ Vt, bf16* __restrict__ O) {
  __shared__ alignas(16) bf16  Ks[32 * ATT_LDK];
  __shared__ alignas(16) bf16  Vs[64 * ATT_LDV];
  __shared__ alignas(16) float Ps[4][16 * ATT_LDP];

  const int lane = threadIdx.x & 31, wave = threadIdx.x >> 5;
  const int bh = blockIdx.y;
  const int b = bh / H_, h = bh % H_;
  const int q0 = blockIdx.x * 64 + wave * 16;
  const int rowStride = H_ * K_;  // 768
  const float scl = 0.125f;       // 1/sqrt(64)

  const bf16* qbase = Q + ((size_t)(b * S_ + q0) * H_ + h) * K_;
  v16bf aq0 = frag_a_global(qbase, rowStride, 0, lane);
  v16bf aq1 = frag_a_global(qbase, rowStride, 32, lane);

  v8f o0 = {}, o1 = {}, o2 = {}, o3 = {};
  float mrun[8], lrun[8];
#pragma unroll
  for (int r = 0; r < 8; ++r) { mrun[r] = -1e30f; lrun[r] = 0.f; }

  for (int kt = 0; kt < S_; kt += 32) {
    __syncthreads();
    // ---- stage K (row-major [key][d], async) and V (transposed [d][key]) ----
    {
      int idx = threadIdx.x * 16;       // 2048 elems / 128 thr
      int key = idx >> 6, d0 = idx & 63;
      const bf16* gk = Kt + ((size_t)(b * S_ + kt + key) * H_ + h) * K_ + d0;
      const bf16* gv = Vt + ((size_t)(b * S_ + kt + key) * H_ + h) * K_ + d0;
#if HAVE_ASYNC_LDS
      async_copy_b128(gk,     &Ks[key * ATT_LDK + d0]);
      async_copy_b128(gk + 8, &Ks[key * ATT_LDK + d0 + 8]);
#else
#pragma unroll
      for (int j = 0; j < 8; ++j)
        *(unsigned*)&Ks[key * ATT_LDK + d0 + 2 * j] =
            *(const unsigned*)(gk + 2 * j);
#endif
#pragma unroll
      for (int j = 0; j < 16; ++j) Vs[(d0 + j) * ATT_LDV + key] = gv[j];
    }
#if HAVE_ASYNC_LDS
    wait_async0();
#endif
    __syncthreads();

    // ---- logits 16x32: S = Q(16x64) * K^T(64x32) ----
    v8f c0 = {}, c1 = {};
    {
      v16bf bk;
      bk = frag_b_lds(Ks, 0,  ATT_LDK, 0,  lane); c0 = wmma_bf16(aq0, bk, c0);
      bk = frag_b_lds(Ks, 0,  ATT_LDK, 32, lane); c0 = wmma_bf16(aq1, bk, c0);
      bk = frag_b_lds(Ks, 16, ATT_LDK, 0,  lane); c1 = wmma_bf16(aq0, bk, c1);
      bk = frag_b_lds(Ks, 16, ATT_LDK, 32, lane); c1 = wmma_bf16(aq1, bk, c1);
    }

    // ---- online softmax (row = r + 8*half, reduce across 16-lane half) ----
    float* P = &Ps[wave][0];
    const int nl = lane & 15, hs = lane >> 4;
#pragma unroll
    for (int r = 0; r < 8; ++r) {
      float t = redmax16(fmaxf(c0[r], c1[r]));
      float mn = fmaxf(mrun[r], t);
      float alpha = __expf((mrun[r] - mn) * scl);
      mrun[r] = mn;
      float p0 = __expf((c0[r] - mn) * scl);
      float p1 = __expf((c1[r] - mn) * scl);
      int m = r + (hs << 3);
      P[m * ATT_LDP + nl]      = p0;
      P[m * ATT_LDP + 16 + nl] = p1;
      float s = redsum16(p0 + p1);
      lrun[r] = lrun[r] * alpha + s;
      o0[r] *= alpha; o1[r] *= alpha; o2[r] *= alpha; o3[r] *= alpha;
    }

    // ---- O += P(16x32) * V(32x64) ----
    v16bf ap = frag_a_f32lds(P, ATT_LDP, lane);
    v16bf bv;
    bv = frag_b_lds(Vs, 0,  ATT_LDV, 0, lane); o0 = wmma_bf16(ap, bv, o0);
    bv = frag_b_lds(Vs, 16, ATT_LDV, 0, lane); o1 = wmma_bf16(ap, bv, o1);
    bv = frag_b_lds(Vs, 32, ATT_LDV, 0, lane); o2 = wmma_bf16(ap, bv, o2);
    bv = frag_b_lds(Vs, 48, ATT_LDV, 0, lane); o3 = wmma_bf16(ap, bv, o3);
  }

  // ---- normalize and write bf16 output [B,S,H,K] ----
  const int nl = lane & 15, hs = lane >> 4;
  bf16* ob = O + ((size_t)(b * S_ + q0) * H_ + h) * K_;
#pragma unroll
  for (int r = 0; r < 8; ++r) {
    float inv = 1.0f / lrun[r];
    bf16* rp = ob + (size_t)(r + (hs << 3)) * rowStride;
    rp[ 0 + nl] = (bf16)(o0[r] * inv);
    rp[16 + nl] = (bf16)(o1[r] * inv);
    rp[32 + nl] = (bf16)(o2[r] * inv);
    rp[48 + nl] = (bf16)(o3[r] * inv);
  }
}

// ---------------------------------------------------------------------------
// Host: orchestrate the full forward pass on `stream`.
// ---------------------------------------------------------------------------
extern "C" void kernel_launch(void* const* d_in, const int* in_sizes, int n_in,
                              void* d_out, int out_size, void* d_ws,
                              size_t ws_size, hipStream_t stream) {
  const float* emb   = (const float*)d_in[0];
  const float* ln1_s = (const float*)d_in[1];
  const float* ln1_b = (const float*)d_in[2];
  const float* wq    = (const float*)d_in[3];
  const float* bq    = (const float*)d_in[4];
  const float* wk    = (const float*)d_in[5];
  const float* bk    = (const float*)d_in[6];
  const float* wv    = (const float*)d_in[7];
  const float* bv    = (const float*)d_in[8];
  const float* wo    = (const float*)d_in[9];
  const float* bo    = (const float*)d_in[10];
  const float* ln2_s = (const float*)d_in[11];
  const float* ln2_b = (const float*)d_in[12];
  const float* w1    = (const float*)d_in[13];
  const float* b1    = (const float*)d_in[14];
  const float* w2    = (const float*)d_in[15];
  const float* b2    = (const float*)d_in[16];
  const float* lnf_s = (const float*)d_in[17];
  const float* lnf_b = (const float*)d_in[18];

  char* p = (char*)d_ws;
  auto alloc = [&](size_t bytes) -> char* {
    char* r = p;
    p += (bytes + 255) & ~(size_t)255;
    return r;
  };
  float* h   = (float*)alloc((size_t)M_ * D_ * 4);
  bf16* hn   = (bf16*)alloc((size_t)M_ * D_ * 2);
  bf16* qb   = (bf16*)alloc((size_t)M_ * D_ * 2);
  bf16* kb_  = (bf16*)alloc((size_t)M_ * D_ * 2);
  bf16* vb_  = (bf16*)alloc((size_t)M_ * D_ * 2);
  bf16* ab   = (bf16*)alloc((size_t)M_ * D_ * 2);
  bf16* hid  = (bf16*)alloc((size_t)M_ * F_ * 2);
  bf16* wqb  = (bf16*)alloc((size_t)L_ * D_ * D_ * 2);
  bf16* wkb  = (bf16*)alloc((size_t)L_ * D_ * D_ * 2);
  bf16* wvb  = (bf16*)alloc((size_t)L_ * D_ * D_ * 2);
  bf16* wob  = (bf16*)alloc((size_t)L_ * D_ * D_ * 2);
  bf16* w1b  = (bf16*)alloc((size_t)L_ * D_ * F_ * 2);
  bf16* w2b  = (bf16*)alloc((size_t)L_ * F_ * D_ * 2);

  const int nWsm = L_ * D_ * D_;  // 1.18M
  const int nWlg = L_ * D_ * F_;  // 4.72M
  cvt_f32_bf16<<<2048, 256, 0, stream>>>(wq, wqb, nWsm);
  cvt_f32_bf16<<<2048, 256, 0, stream>>>(wk, wkb, nWsm);
  cvt_f32_bf16<<<2048, 256, 0, stream>>>(wv, wvb, nWsm);
  cvt_f32_bf16<<<2048, 256, 0, stream>>>(wo, wob, nWsm);
  cvt_f32_bf16<<<4096, 256, 0, stream>>>(w1, w1b, nWlg);
  cvt_f32_bf16<<<4096, 256, 0, stream>>>(w2, w2b, nWlg);
  copy_f32<<<4096, 256, 0, stream>>>(emb, h, M_ * D_);

  const dim3 gD(D_ / TN, M_ / TM);   // GEMMs with N = 768
  const dim3 gF(F_ / TN, M_ / TM);   // GEMM with N = 3072
  const dim3 gAtt(S_ / 64, B_ * H_);

  for (int i = 0; i < L_; ++i) {
    layernorm_k<0><<<M_, 256, 0, stream>>>(h, ln1_s + i * D_, ln1_b + i * D_,
                                           hn, nullptr);
    gemm_bf16_wmma<EP_BF16_BIAS><<<gD, 256, 0, stream>>>(
        hn, wqb + (size_t)i * D_ * D_, bq + i * D_, nullptr, qb, nullptr, D_, D_);
    gemm_bf16_wmma<EP_BF16_BIAS><<<gD, 256, 0, stream>>>(
        hn, wkb + (size_t)i * D_ * D_, bk + i * D_, nullptr, kb_, nullptr, D_, D_);
    gemm_bf16_wmma<EP_BF16_BIAS><<<gD, 256, 0, stream>>>(
        hn, wvb + (size_t)i * D_ * D_, bv + i * D_, nullptr, vb_, nullptr, D_, D_);
    attn_flash<<<gAtt, 128, 0, stream>>>(qb, kb_, vb_, ab);
    gemm_bf16_wmma<EP_F32_BIAS_RES><<<gD, 256, 0, stream>>>(
        ab, wob + (size_t)i * D_ * D_, bo + i * D_, h, nullptr, h, D_, D_);
    layernorm_k<0><<<M_, 256, 0, stream>>>(h, ln2_s + i * D_, ln2_b + i * D_,
                                           hn, nullptr);
    gemm_bf16_wmma<EP_BF16_BIAS_GELU><<<gF, 256, 0, stream>>>(
        hn, w1b + (size_t)i * D_ * F_, b1 + i * F_, nullptr, hid, nullptr, F_, D_);
    gemm_bf16_wmma<EP_F32_BIAS_RES><<<gD, 256, 0, stream>>>(
        hid, w2b + (size_t)i * F_ * D_, b2 + i * D_, h, nullptr, h, D_, F_);
  }
  layernorm_k<1><<<M_, 256, 0, stream>>>(h, lnf_s, lnf_b, nullptr,
                                         (float*)d_out);
}
